// ScaledDotProductAttention_25460566131168
// MI455X (gfx1250) — matrix operand order repaired and checked by reading, not compiled, not run on hardware
//
#include <hip/hip_runtime.h>

typedef __attribute__((ext_vector_type(16))) __bf16 v16bf;
typedef __attribute__((ext_vector_type(8)))  __bf16 v8bf;
typedef __attribute__((ext_vector_type(2)))  __bf16 v2bf;
typedef __attribute__((ext_vector_type(8)))  float  v8f;
typedef __attribute__((ext_vector_type(4)))  float  v4f;

namespace {
constexpr int kB = 4, kH = 16, kS = 2048, kD = 64;
constexpr int QTILE  = 128;          // queries per workgroup (8 waves x 16)
constexpr int KTILE  = 64;           // keys per inner iteration
constexpr int NWAVES = QTILE / 16;   // 8 waves = 256 threads (wave32)
constexpr int KLD = kD + 8;          // Kl row stride in bf16 elems (144B, 16B-aligned)
constexpr int VLD = KTILE + 8;       // Vt row stride (144B, 16B-aligned)
constexpr int PLD = KTILE + 8;       // P  row stride (144B, 16B-aligned)
}

// XOR-butterfly within 16-lane halves via ds_swizzle (group-of-32 encoding:
// offset[14:10]=xor, [9:5]=or, [4:0]=and). Masks < 16 never cross the half.
template<int MASK>
__device__ __forceinline__ float swz_xor_f(float x) {
  return __int_as_float(
      __builtin_amdgcn_ds_swizzle(__float_as_int(x), (MASK << 10) | 0x1f));
}

__device__ __forceinline__ v16bf join16(v8bf lo, v8bf hi) {
  v16bf r;
#pragma unroll
  for (int e = 0; e < 8; ++e) { r[e] = lo[e]; r[8 + e] = hi[e]; }
  return r;
}

// B-fragment (K=32 x N=16 bf16): per lane two 8-element runs at p and p+16 (split layout).
__device__ __forceinline__ v16bf ldsB(const __bf16* p) {
  return join16(*(const v8bf*)p, *(const v8bf*)(p + 16));
}

__device__ __forceinline__ v8f wmma_bf16(v16bf a, v16bf b, v8f c) {
  return __builtin_amdgcn_wmma_f32_16x16x32_bf16(false, a, false, b,
                                                 (short)0, c, false, false);
}

__global__ void __launch_bounds__(256) fa_fwd(const float* __restrict__ Q,
                                              const float* __restrict__ K,
                                              const float* __restrict__ V,
                                              float* __restrict__ O) {
  __shared__ alignas(16) __bf16 Kl[KTILE][KLD];        // keys row-major
  __shared__ alignas(16) __bf16 Vt[kD][VLD];           // V transposed (dim-major)
  __shared__ alignas(16) __bf16 Pb[NWAVES][16][PLD];   // wave-private P staging

  const int tid  = threadIdx.x;
  const int lane = tid & 31;
  const int wave = tid >> 5;
  const int m    = lane & 15;
  const int half = lane >> 4;

  // Reverse qb order: heaviest (largest-qb) causal blocks launch first (LPT).
  const int qb = (kS / QTILE - 1 - blockIdx.x) * QTILE;
  const int bh = blockIdx.y;
  const int qw = qb + wave * 16;          // first query row of this wave

  const size_t base = (size_t)bh * kS * kD;
  const float* Qg = Q + base;
  const float* Kg = K + base;
  const float* Vg = V + base;
  float*       Og = O + base;

  // ---- Q A-fragments (split layout), pre-scaled by log2(e)/sqrt(D) ----
  // Scores live in the log2 domain so softmax uses v_exp_f32 (exp2) directly.
  constexpr float scale = 0.125f * 1.44269504088896340736f;
  v16bf Aq[2];
  {
    const float* qp = Qg + (size_t)(qw + m) * kD + 8 * half;
#pragma unroll
    for (int kc = 0; kc < 2; ++kc) {     // d-chunks [0,32) and [32,64)
      const float* p0 = qp + kc * 32;
      v16bf a;
#pragma unroll
      for (int e = 0; e < 8; ++e) a[e]     = (__bf16)(p0[e] * scale);
#pragma unroll
      for (int e = 0; e < 8; ++e) a[8 + e] = (__bf16)(p0[16 + e] * scale);
      Aq[kc] = a;
    }
  }

  v8f   Oacc[4] = {};
  float m_i[8], l_i[8];
#pragma unroll
  for (int r = 0; r < 8; ++r) { m_i[r] = -1e30f; l_i[r] = 0.f; }

  // Stager mapping: each thread owns 2 adjacent keys x 8 dims.
  const int kp2 = (tid >> 3) * 2;   // key pair base (0,2,..,62)
  const int ld0 = (tid & 7) * 8;    // dim start (0..56)

  const int nkb = (qb + QTILE) / KTILE;

  // ---- software pipeline: global loads for tile kb+1 overlap compute of kb ----
  v4f kreg[4], vreg[4];
  auto load_tile = [&](int kb) {
    const float* kptr = Kg + (size_t)(kb * KTILE + kp2) * kD + ld0;
    kreg[0] = *(const v4f*)kptr;
    kreg[1] = *(const v4f*)(kptr + 4);
    kreg[2] = *(const v4f*)(kptr + kD);
    kreg[3] = *(const v4f*)(kptr + kD + 4);
    const float* vptr = Vg + (size_t)(kb * KTILE + kp2) * kD + ld0;
    vreg[0] = *(const v4f*)vptr;
    vreg[1] = *(const v4f*)(vptr + 4);
    vreg[2] = *(const v4f*)(vptr + kD);
    vreg[3] = *(const v4f*)(vptr + kD + 4);
  };
  auto store_tile = [&]() {
    v8bf pk0, pk1;
#pragma unroll
    for (int e = 0; e < 4; ++e) {
      pk0[e] = (__bf16)kreg[0][e]; pk0[4 + e] = (__bf16)kreg[1][e];
      pk1[e] = (__bf16)kreg[2][e]; pk1[4 + e] = (__bf16)kreg[3][e];
    }
    *(v8bf*)&Kl[kp2][ld0]     = pk0;
    *(v8bf*)&Kl[kp2 + 1][ld0] = pk1;
#pragma unroll
    for (int e = 0; e < 4; ++e) {   // transposed, packed key-pair stores
      v2bf lo; lo[0] = (__bf16)vreg[0][e]; lo[1] = (__bf16)vreg[2][e];
      v2bf hi; hi[0] = (__bf16)vreg[1][e]; hi[1] = (__bf16)vreg[3][e];
      *(v2bf*)&Vt[ld0 + e][kp2]     = lo;
      *(v2bf*)&Vt[ld0 + 4 + e][kp2] = hi;
    }
  };

  load_tile(0);
  for (int kb = 0; kb < nkb; ++kb) {
    store_tile();
    __syncthreads();
    if (kb + 1 < nkb) load_tile(kb + 1);   // next tile rides under compute

    if (kb * KTILE <= qw + 15) {   // wave-uniform: any unmasked work here?
      // ---- S = Q K^T : 16 queries x 64 keys (8 WMMA) ----
      v8f C[4] = {};
#pragma unroll
      for (int kc = 0; kc < 2; ++kc) {
        v16bf Bk[4];
#pragma unroll
        for (int nf = 0; nf < 4; ++nf)
          Bk[nf] = ldsB(&Kl[nf * 16 + m][kc * 32 + 8 * half]);
#pragma unroll
        for (int nf = 0; nf < 4; ++nf)
          C[nf] = wmma_bf16(Aq[kc], Bk[nf], C[nf]);
      }

      // ---- causal mask (diagonal blocks only) ----
      const int krow0 = kb * KTILE;
      if (krow0 + KTILE - 1 > qw) {
#pragma unroll
        for (int nf = 0; nf < 4; ++nf) {
          const int key = krow0 + nf * 16 + m;
#pragma unroll
          for (int r = 0; r < 8; ++r) {
            if (key > qw + r + 8 * half) C[nf][r] = -1e30f;
          }
        }
      }

      // ---- online softmax (log2 domain) in C-fragment layout ----
#pragma unroll
      for (int r = 0; r < 8; ++r) {
        float mx = fmaxf(fmaxf(C[0][r], C[1][r]), fmaxf(C[2][r], C[3][r]));
        mx = fmaxf(mx, swz_xor_f<1>(mx));
        mx = fmaxf(mx, swz_xor_f<2>(mx));
        mx = fmaxf(mx, swz_xor_f<4>(mx));
        mx = fmaxf(mx, swz_xor_f<8>(mx));
        const float mn   = fmaxf(m_i[r], mx);
        const float corr = __builtin_amdgcn_exp2f(m_i[r] - mn);
        m_i[r] = mn;
        float p[4];
#pragma unroll
        for (int nf = 0; nf < 4; ++nf)
          p[nf] = __builtin_amdgcn_exp2f(C[nf][r] - mn);
        l_i[r] = l_i[r] * corr + ((p[0] + p[1]) + (p[2] + p[3]));
#pragma unroll
        for (int f = 0; f < 4; ++f) Oacc[f][r] *= corr;
        const int prow = r + 8 * half;
#pragma unroll
        for (int nf = 0; nf < 4; ++nf)
          Pb[wave][prow][nf * 16 + m] = (__bf16)p[nf];
      }
      __builtin_amdgcn_wave_barrier();      // same-wave DS ops are in-order in HW

      // ---- reload P as two A-fragments (split layout, keys [0,32),[32,64)) ----
      const __bf16* pp = &Pb[wave][m][8 * half];
      v16bf Ap0 = join16(*(const v8bf*)pp,        *(const v8bf*)(pp + 16));
      v16bf Ap1 = join16(*(const v8bf*)(pp + 32), *(const v8bf*)(pp + 48));

      // ---- O += P V : 16 x 64, K=64 (8 WMMA) ----
#pragma unroll
      for (int f = 0; f < 4; ++f) {
        const __bf16* vb = &Vt[16 * f + m][8 * half];
        v16bf Bv0 = ldsB(vb);
        v16bf Bv1 = ldsB(vb + 32);
        Oacc[f] = wmma_bf16(Ap0, Bv0, Oacc[f]);
        Oacc[f] = wmma_bf16(Ap1, Bv1, Oacc[f]);
      }
    }
    __syncthreads();
  }

  // ---- finalize: full row sums, normalize, store ----
#pragma unroll
  for (int r = 0; r < 8; ++r) {
    float l = l_i[r];
    l += swz_xor_f<1>(l);
    l += swz_xor_f<2>(l);
    l += swz_xor_f<4>(l);
    l += swz_xor_f<8>(l);
    const float inv = 1.0f / l;
    float* op = Og + (size_t)(qw + r + 8 * half) * kD + m;
#pragma unroll
    for (int f = 0; f < 4; ++f) op[16 * f] = Oacc[f][r] * inv;
  }
}

extern "C" void kernel_launch(void* const* d_in, const int* in_sizes, int n_in,
                              void* d_out, int out_size, void* d_ws, size_t ws_size,
                              hipStream_t stream) {
  (void)in_sizes; (void)n_in; (void)out_size; (void)d_ws; (void)ws_size;
  const float* q = (const float*)d_in[0];
  const float* k = (const float*)d_in[1];
  const float* v = (const float*)d_in[2];
  // d_in[3] is the causal mask; causality is computed analytically in-kernel.
  float* out = (float*)d_out;
  dim3 grid(kS / QTILE, kB * kH);
  fa_fwd<<<grid, dim3(256), 0, stream>>>(q, k, v, out);
}
